// CausalSelfAttention_26139170963952
// MI455X (gfx1250) — matrix operand order repaired and checked by reading, not compiled, not run on hardware
//
#include <hip/hip_runtime.h>
#include <math.h>

typedef __attribute__((ext_vector_type(16))) _Float16 v16h;
typedef __attribute__((ext_vector_type(8)))  _Float16 v8h;
typedef __attribute__((ext_vector_type(4)))  _Float16 v4h;
typedef __attribute__((ext_vector_type(8)))  float    v8f;
typedef __attribute__((ext_vector_type(4)))  float    v4f;
typedef __attribute__((ext_vector_type(4)))  int      v4i;

#define EMB_   2048
#define HEADS_ 16
#define HEAD_  128
#define BB_    4
#define SS_    2048

union AF { v16h v; v8h h[2]; };

__device__ inline v4h cvt4(v4f d) {
    v4h h;
    h[0] = (_Float16)d[0]; h[1] = (_Float16)d[1];
    h[2] = (_Float16)d[2]; h[3] = (_Float16)d[3];
    return h;
}

// ---- CDNA5 async global->LDS copy (ASYNCcnt-tracked), with safe fallback ----
#if defined(__gfx1250__) && __has_builtin(__builtin_amdgcn_global_load_async_to_lds_b128)
#define HAVE_ASYNC_LDS 1
#else
#define HAVE_ASYNC_LDS 0
#endif

__device__ inline void copy16B(_Float16* lds_dst, const _Float16* g_src) {
#if HAVE_ASYNC_LDS
    __builtin_amdgcn_global_load_async_to_lds_b128(
        (__attribute__((address_space(1))) v4i*)g_src,
        (__attribute__((address_space(3))) v4i*)lds_dst,
        0, 0);
#else
    *(v8h*)lds_dst = *(const v8h*)g_src;
#endif
}

__device__ inline void wait_async0() {
#if HAVE_ASYNC_LDS
#if __has_builtin(__builtin_amdgcn_s_wait_asynccnt)
    __builtin_amdgcn_s_wait_asynccnt(0);
#else
    asm volatile("s_wait_asynccnt 0" ::: "memory");
#endif
#endif
}

// ---------------------------------------------------------------------------
// Kernel 1: fused QKV projection.  y = x @ W^T + b  (torch Linear semantics)
// Register-staged double buffering; next-tile global loads are issued BEFORE
// the publish barrier so they overlap the barrier-wait + current-tile WMMAs.
// Q,K stored [B,H,S,D]; V stored transposed [B,H,D,S].
// ---------------------------------------------------------------------------
__launch_bounds__(256)
__global__ void qkv_proj_kernel(const float* __restrict__ x,
                                const float* __restrict__ Wq,
                                const float* __restrict__ Wk,
                                const float* __restrict__ Wv,
                                const float* __restrict__ bq,
                                const float* __restrict__ bk,
                                const float* __restrict__ bv,
                                _Float16* __restrict__ Qh,
                                _Float16* __restrict__ Kh,
                                _Float16* __restrict__ Vt)
{
    __shared__ _Float16 As[128 * 32];
    __shared__ _Float16 Bs[128 * 32];

    const int z = blockIdx.z;
    const float* W    = (z == 0) ? Wq : (z == 1) ? Wk : Wv;
    const float* bias = (z == 0) ? bq : (z == 1) ? bk : bv;

    const int tid  = threadIdx.x;
    const int lane = tid & 31;
    const int w    = tid >> 5;
    const int lm   = lane & 15;
    const int hf   = lane >> 4;
    const int wm   = w & 1;        // 2 M-groups of 64 rows
    const int wn   = w >> 1;       // 4 N-groups of 32 cols

    const int m0 = blockIdx.y * 128;
    const int n0 = blockIdx.x * 128;

    v8f zero = {};
    v8f acc[4][2];
    for (int i = 0; i < 4; i++)
        for (int j = 0; j < 2; j++) acc[i][j] = zero;

    v4f ra[4], rb[4];
    auto loadTile = [&](int k0) {
        for (int i = 0; i < 4; i++) {
            int idx = tid + i * 256;       // 1024 float4 slots per operand
            int row = idx >> 3;
            int c4  = idx & 7;
            ra[i] = *(const v4f*)(x + (size_t)(m0 + row) * EMB_ + k0 + c4 * 4);
            rb[i] = *(const v4f*)(W + (size_t)(n0 + row) * EMB_ + k0 + c4 * 4);
        }
    };

    loadTile(0);
    for (int kt = 0; kt < EMB_ / 32; kt++) {
        __syncthreads();                       // previous tile's reads done
        for (int i = 0; i < 4; i++) {
            int idx = tid + i * 256;
            int row = idx >> 3;
            int c4  = idx & 7;
            *(v4h*)&As[row * 32 + c4 * 4] = cvt4(ra[i]);
            *(v4h*)&Bs[row * 32 + c4 * 4] = cvt4(rb[i]);
        }
        // issue next tile's global loads before the barrier: the fence pins
        // them here, so their latency hides under barrier-wait + WMMAs
        if (kt < EMB_ / 32 - 1) loadTile((kt + 1) * 32);
        __syncthreads();                       // tile visible

        AF a[4], b[2];
        for (int mt = 0; mt < 4; mt++) {
            const _Float16* p = &As[(wm * 64 + mt * 16 + lm) * 32];
            a[mt].h[0] = *(const v8h*)(p + hf * 8);
            a[mt].h[1] = *(const v8h*)(p + 16 + hf * 8);
        }
        for (int nt = 0; nt < 2; nt++) {
            const _Float16* p = &Bs[(wn * 32 + nt * 16 + lm) * 32];
            b[nt].h[0] = *(const v8h*)(p + hf * 8);
            b[nt].h[1] = *(const v8h*)(p + 16 + hf * 8);
        }
        for (int mt = 0; mt < 4; mt++)
            for (int nt = 0; nt < 2; nt++)
                acc[mt][nt] = __builtin_amdgcn_wmma_f32_16x16x32_f16(
                    false, a[mt].v, false, b[nt].v, (short)0, acc[mt][nt],
                    false, false);
    }

    for (int mt = 0; mt < 4; mt++)
        for (int nt = 0; nt < 2; nt++) {
            int n = n0 + wn * 32 + nt * 16 + lm;
            int h = n >> 7, d = n & 127;
            float bv_ = bias[n];
            for (int r = 0; r < 8; r++) {
                int m = m0 + wm * 64 + mt * 16 + r + hf * 8;
                int bidx = m >> 11, s = m & 2047;
                _Float16 hv = (_Float16)(acc[mt][nt][r] + bv_);
                if (z == 2) {
                    Vt[(((size_t)bidx * HEADS_ + h) * HEAD_ + d) * SS_ + s] = hv;
                } else {
                    _Float16* dst = (z == 0) ? Qh : Kh;
                    dst[(((size_t)bidx * HEADS_ + h) * SS_ + s) * HEAD_ + d] = hv;
                }
            }
        }
}

// ---------------------------------------------------------------------------
// Kernel 2: flash attention, one (b,h) x 128-query block per workgroup.
// Double-buffered K/V staging via async global->LDS DMA, dedicated P region:
//   LDS = 2 x (Ks 32KB + Vs 32KB) + Ps 32KB = 160 KB (2 WGs per 320KB WGP).
// One s_wait_asynccnt + one barrier per key block; prefetch overlaps compute.
// ---------------------------------------------------------------------------
__launch_bounds__(256)
__global__ void attn_kernel(const _Float16* __restrict__ Qh,
                            const _Float16* __restrict__ Kh,
                            const _Float16* __restrict__ Vt,
                            _Float16* __restrict__ ctx)
{
    extern __shared__ _Float16 smem[];
    // smem: [buf0: Ks 16384 | Vs 16384][buf1: Ks | Vs][Ps 8*2048]

    const int tid  = threadIdx.x;
    const int lane = tid & 31;
    const int w    = tid >> 5;
    const int lm   = lane & 15;
    const int hf   = lane >> 4;

    const int qb = blockIdx.x;
    const int bh = blockIdx.y;
    const size_t qkBase = (size_t)bh * SS_ * HEAD_;
    const size_t vBase  = (size_t)bh * HEAD_ * SS_;
    const int q0 = qb * 128;

    const int crow = tid >> 1;            // staging: 2 threads per 128-h row
    const int cc0  = (tid & 1) * 64;

    auto stage = [&](int pb, int kb) {
        _Float16* Kd = smem + pb * 32768 + crow * 128 + cc0;
        _Float16* Vd = smem + pb * 32768 + 16384 + crow * 128 + cc0;
        const _Float16* gK = Kh + qkBase + (size_t)(kb * 128 + crow) * HEAD_ + cc0;
        const _Float16* gV = Vt + vBase + (size_t)crow * SS_ + kb * 128 + cc0;
        for (int i = 0; i < 8; i++) {
            copy16B(Kd + i * 8, gK + i * 8);
            copy16B(Vd + i * 8, gV + i * 8);
        }
    };

    // Q A-fragments for this wave's 16 rows, 4 chunks over D
    AF qf[4];
    {
        const int qrow = q0 + w * 16 + lm;
        for (int kk = 0; kk < 4; kk++) {
            const _Float16* p = Qh + qkBase + (size_t)qrow * HEAD_ + kk * 32;
            qf[kk].h[0] = *(const v8h*)(p + hf * 8);
            qf[kk].h[1] = *(const v8h*)(p + 16 + hf * 8);
        }
    }

    v8f zero = {};
    v8f O[8];
    for (int t = 0; t < 8; t++) O[t] = zero;
    float mrun[8], lrun[8];
    for (int r = 0; r < 8; r++) { mrun[r] = -3.0e38f; lrun[r] = 0.0f; }

    const float scale = 0.08838834764831845f;   // 1/sqrt(128)

    stage(0, 0);                                 // prologue prefetch
    for (int kb = 0; kb <= qb; kb++) {
        const int cur = kb & 1;
        wait_async0();        // my async issues for buf[cur] complete
        __syncthreads();      // everyone's complete -> buf[cur] published
        if (kb < qb) stage(1 - cur, kb + 1);     // prefetch overlaps compute

        const _Float16* Ks = smem + cur * 32768;
        const _Float16* Vs = Ks + 16384;

        // scores S = Q K^T : 8 key tiles of 16, K-dim = D = 128
        v8f sc[8];
        for (int t = 0; t < 8; t++) {
            v8f a = zero;
            for (int kk = 0; kk < 4; kk++) {
                AF b;
                const _Float16* p = &Ks[(t * 16 + lm) * 128 + kk * 32];
                b.h[0] = *(const v8h*)(p + hf * 8);
                b.h[1] = *(const v8h*)(p + 16 + hf * 8);
                a = __builtin_amdgcn_wmma_f32_16x16x32_f16(
                        false, qf[kk].v, false, b.v, (short)0, a, false, false);
            }
            sc[t] = a;
        }

        // scale + causal mask (diagonal block only)
        const bool diag = (kb == qb);
        for (int t = 0; t < 8; t++) {
            int key = kb * 128 + t * 16 + lm;
            for (int r = 0; r < 8; r++) {
                float s = sc[t][r] * scale;
                if (diag) {
                    int q = q0 + w * 16 + r + hf * 8;
                    if (key > q) s = -1.0e30f;
                }
                sc[t][r] = s;
            }
        }

        // online softmax (rows live across the 16-lane N group)
        for (int r = 0; r < 8; r++) {
            float bm = sc[0][r];
            for (int t = 1; t < 8; t++) bm = fmaxf(bm, sc[t][r]);
            for (int off = 8; off > 0; off >>= 1)
                bm = fmaxf(bm, __shfl_xor(bm, off, 16));
            float nm    = fmaxf(mrun[r], bm);
            float alpha = __expf(mrun[r] - nm);
            mrun[r] = nm;
            float rs = 0.0f;
            for (int t = 0; t < 8; t++) {
                float p = __expf(sc[t][r] - nm);
                sc[t][r] = p;
                rs += p;
            }
            for (int off = 8; off > 0; off >>= 1)
                rs += __shfl_xor(rs, off, 16);
            lrun[r] = lrun[r] * alpha + rs;
            for (int t = 0; t < 8; t++) O[t][r] *= alpha;
        }

        // P -> dedicated wave-private LDS patch (no barrier needed: LDS ops
        // from the same wave are in-order)
        _Float16* Ps = smem + 65536 + w * (16 * 128);
        for (int t = 0; t < 8; t++)
            for (int r = 0; r < 8; r++)
                Ps[(r + hf * 8) * 128 + t * 16 + lm] = (_Float16)sc[t][r];

        AF pa[4];
        for (int kk = 0; kk < 4; kk++) {
            const _Float16* pp = Ps + lm * 128 + kk * 32;
            pa[kk].h[0] = *(const v8h*)(pp + hf * 8);
            pa[kk].h[1] = *(const v8h*)(pp + 16 + hf * 8);
        }
        for (int t = 0; t < 8; t++) {
            for (int kk = 0; kk < 4; kk++) {
                AF vb;
                const _Float16* vp = &Vs[(t * 16 + lm) * 128 + kk * 32];
                vb.h[0] = *(const v8h*)(vp + hf * 8);
                vb.h[1] = *(const v8h*)(vp + 16 + hf * 8);
                O[t] = __builtin_amdgcn_wmma_f32_16x16x32_f16(
                           false, pa[kk].v, false, vb.v, (short)0, O[t],
                           false, false);
            }
        }
    }

    // normalize and store ctx [B,S,H*D] f16
    const int b = bh >> 4, h = bh & 15;
    for (int r = 0; r < 8; r++) {
        float inv = 1.0f / lrun[r];
        int s = q0 + w * 16 + r + hf * 8;
        size_t rowBase = ((size_t)b * SS_ + s) * EMB_ + (size_t)h * HEAD_;
        for (int t = 0; t < 8; t++)
            ctx[rowBase + t * 16 + lm] = (_Float16)(O[t][r] * inv);
    }
}

// ---------------------------------------------------------------------------
// Kernel 3: output projection  out = ctx @ Wo^T + bo   (f32 output),
// register-staged double buffering like kernel 1.
// ---------------------------------------------------------------------------
__launch_bounds__(256)
__global__ void out_proj_kernel(const _Float16* __restrict__ ctx,
                                const float* __restrict__ Wo,
                                const float* __restrict__ bo,
                                float* __restrict__ out)
{
    __shared__ _Float16 As[128 * 32];
    __shared__ _Float16 Bs[128 * 32];

    const int tid  = threadIdx.x;
    const int lane = tid & 31;
    const int w    = tid >> 5;
    const int lm   = lane & 15;
    const int hf   = lane >> 4;
    const int wm   = w & 1;
    const int wn   = w >> 1;

    const int m0 = blockIdx.y * 128;
    const int n0 = blockIdx.x * 128;

    v8f zero = {};
    v8f acc[4][2];
    for (int i = 0; i < 4; i++)
        for (int j = 0; j < 2; j++) acc[i][j] = zero;

    const int arow = tid >> 1;
    const int ac0  = (tid & 1) * 16;
    v8h rA[2];
    v4f rb[4];
    auto loadTile = [&](int k0) {
        const _Float16* g = ctx + (size_t)(m0 + arow) * EMB_ + k0 + ac0;
        rA[0] = *(const v8h*)g;
        rA[1] = *(const v8h*)(g + 8);
        for (int i = 0; i < 4; i++) {
            int idx = tid + i * 256;
            int row = idx >> 3;
            int c4  = idx & 7;
            rb[i] = *(const v4f*)(Wo + (size_t)(n0 + row) * EMB_ + k0 + c4 * 4);
        }
    };

    loadTile(0);
    for (int kt = 0; kt < EMB_ / 32; kt++) {
        __syncthreads();
        *(v8h*)&As[arow * 32 + ac0]     = rA[0];
        *(v8h*)&As[arow * 32 + ac0 + 8] = rA[1];
        for (int i = 0; i < 4; i++) {
            int idx = tid + i * 256;
            int row = idx >> 3;
            int c4  = idx & 7;
            *(v4h*)&Bs[row * 32 + c4 * 4] = cvt4(rb[i]);
        }
        if (kt < EMB_ / 32 - 1) loadTile((kt + 1) * 32);   // pre-barrier issue
        __syncthreads();

        AF a[4], b[2];
        for (int mt = 0; mt < 4; mt++) {
            const _Float16* p = &As[(wm * 64 + mt * 16 + lm) * 32];
            a[mt].h[0] = *(const v8h*)(p + hf * 8);
            a[mt].h[1] = *(const v8h*)(p + 16 + hf * 8);
        }
        for (int nt = 0; nt < 2; nt++) {
            const _Float16* p = &Bs[(wn * 32 + nt * 16 + lm) * 32];
            b[nt].h[0] = *(const v8h*)(p + hf * 8);
            b[nt].h[1] = *(const v8h*)(p + 16 + hf * 8);
        }
        for (int mt = 0; mt < 4; mt++)
            for (int nt = 0; nt < 2; nt++)
                acc[mt][nt] = __builtin_amdgcn_wmma_f32_16x16x32_f16(
                    false, a[mt].v, false, b[nt].v, (short)0, acc[mt][nt],
                    false, false);
    }

    for (int mt = 0; mt < 4; mt++)
        for (int nt = 0; nt < 2; nt++) {
            int n = n0 + wn * 32 + nt * 16 + lm;
            float bv_ = bo[n];
            for (int r = 0; r < 8; r++) {
                int m = m0 + wm * 64 + mt * 16 + r + hf * 8;
                out[(size_t)m * EMB_ + n] = acc[mt][nt][r] + bv_;
            }
        }
}

// ---------------------------------------------------------------------------
extern "C" void kernel_launch(void* const* d_in, const int* in_sizes, int n_in,
                              void* d_out, int out_size, void* d_ws, size_t ws_size,
                              hipStream_t stream)
{
    (void)in_sizes; (void)n_in; (void)out_size; (void)ws_size;

    const float* x  = (const float*)d_in[0];
    // d_in[1] = attn_mask: causal mask recomputed analytically
    const float* Wq = (const float*)d_in[2];
    const float* bq = (const float*)d_in[3];
    const float* Wk = (const float*)d_in[4];
    const float* bk = (const float*)d_in[5];
    const float* Wv = (const float*)d_in[6];
    const float* bv = (const float*)d_in[7];
    const float* Wo = (const float*)d_in[8];
    const float* bo = (const float*)d_in[9];
    float* out = (float*)d_out;

    const size_t perTensor = (size_t)BB_ * HEADS_ * SS_ * HEAD_;  // 16.78M halves
    _Float16* Qh  = (_Float16*)d_ws;
    _Float16* Kh  = Qh + perTensor;
    _Float16* Vt  = Kh + perTensor;
    _Float16* ctx = Vt + perTensor;

    dim3 blk(256);
    qkv_proj_kernel<<<dim3(EMB_ / 128, (BB_ * SS_) / 128, 3), blk, 0, stream>>>(
        x, Wq, Wk, Wv, bq, bk, bv, Qh, Kh, Vt);

    size_t attnLds = (size_t)(2 * 32768 + 8 * 16 * 128) * sizeof(_Float16); // 160 KB
    attn_kernel<<<dim3(SS_ / 128, BB_ * HEADS_), blk, attnLds, stream>>>(
        Qh, Kh, Vt, ctx);

    out_proj_kernel<<<dim3(EMB_ / 128, (BB_ * SS_) / 128), blk, 0, stream>>>(
        ctx, Wo, bo, out);
}
